// TaggingFNNDecoder_49632642072793
// MI455X (gfx1250) — matrix operand order repaired and checked by reading, not compiled, not run on hardware
//
#include <hip/hip_runtime.h>
#include <hip/hip_bf16.h>
#include <math.h>

// Problem sizes
#define NWG   16
#define TPB   256
#define Bz    32
#define Tz    256
#define Dz    1024
#define Hz    512
#define Kz    128
#define DKz   1152      // D + K
#define G4z   2048      // 4*H
// LDS strides (element units), padded so row stride % 64 banks == 4 dwords
#define XST   1032      // x double buffer [32][1024] bf16, stride 1032
#define SST   136       // sm buffer [32][128] bf16, stride 136
#define HST   520       // h1/h2 [32][512] bf16, stride 520
#define GST   132       // gates [32][128] f32, stride 132
// bytes: 2*32*1032*2 + 32*136*2 + 2*32*520*2 + 32*132*4 + 2*1024*4 + 32*8*4
#define LDS_BYTES 233472

typedef __attribute__((ext_vector_type(16))) __bf16 v16bf;
typedef __attribute__((ext_vector_type(8)))  float  v8f;

union FragBF {
  v16bf v;
  int4  q[2];
};

// ---------------------------------------------------------------------------
// One wave computes a 16-col N-tile for BOTH 16-row M-tiles (B=32), over the
// K-tile sub-range [kt0, kt0+nkt) of a matrix whose full K has KTtot tiles.
// B matrix pre-packed: per (ntile,ktile) a 1024-byte blob; lane l owns 32
// contiguous bytes: k = kt*32 + (l>>4)*16 + e , n = nt*16 + (l&15).
// A matrix row-major bf16 in LDS: lane l reads two 16B chunks of row (l&15)
// [+16 rows for M-tile 1] at k-offsets ((l>>4)*16) and +8.
// ---------------------------------------------------------------------------
__device__ __forceinline__ void wave_gemm(const __bf16* __restrict__ Bpk,
                                          int ntile, int KTtot, int kt0, int nkt,
                                          const __bf16* A, int astr, int lane,
                                          v8f& acc0, v8f& acc1)
{
  const int l15 = lane & 15;
  const int lhi = (lane >> 4) & 1;
  const int4* bp = (const int4*)Bpk + ((size_t)ntile * KTtot + kt0) * 64 + lane * 2;
  const __bf16* ar0 = A + (size_t)l15 * astr + lhi * 16;
  const __bf16* ar1 = ar0 + (size_t)16 * astr;
  for (int kt = 0; kt < nkt; ++kt) {
    FragBF bfr, afr0, afr1;
    bfr.q[0] = bp[0];
    bfr.q[1] = bp[1];
    __builtin_prefetch(bp + 64, 0, 1);   // global_prefetch next K-tile of weights
    bp += 64;
    const __bf16* p0 = ar0 + kt * 32;
    afr0.q[0] = *(const int4*)(p0);
    afr0.q[1] = *(const int4*)(p0 + 8);
    const __bf16* p1 = ar1 + kt * 32;
    afr1.q[0] = *(const int4*)(p1);
    afr1.q[1] = *(const int4*)(p1 + 8);
    acc0 = __builtin_amdgcn_wmma_f32_16x16x32_bf16(false, afr0.v, false, bfr.v,
                                                   (short)0, acc0, false, false);
    acc1 = __builtin_amdgcn_wmma_f32_16x16x32_bf16(false, afr1.v, false, bfr.v,
                                                   (short)0, acc1, false, false);
  }
}

// C/D layout: lane l, vgpr v holds (M = mt*16 + v + (l>=16)*8, N = nt*16 + (l&15))
__device__ __forceinline__ void store_tile(float* gbuf, int colbase, v8f a,
                                           int mt, int lane)
{
  int row0 = mt * 16 + ((lane >> 4) & 1) * 8;
  int col  = colbase + (lane & 15);
  #pragma unroll
  for (int v = 0; v < 8; ++v) gbuf[(row0 + v) * GST + col] = a[v];
}

__device__ __forceinline__ float sigf(float x) { return 1.f / (1.f + __expf(-x)); }

// Sense-reversing grid barrier across NWG persistent workgroups (L2 atomics).
__device__ __forceinline__ void grid_barrier(unsigned* bar)
{
  __syncthreads();
  if (threadIdx.x == 0) {
    __threadfence();
    unsigned gen = __hip_atomic_load(&bar[1], __ATOMIC_ACQUIRE, __HIP_MEMORY_SCOPE_AGENT);
    unsigned arrived = __hip_atomic_fetch_add(&bar[0], 1u, __ATOMIC_ACQ_REL,
                                              __HIP_MEMORY_SCOPE_AGENT);
    if (arrived == (unsigned)(NWG - 1)) {
      __hip_atomic_store(&bar[0], 0u, __ATOMIC_RELAXED, __HIP_MEMORY_SCOPE_AGENT);
      __hip_atomic_fetch_add(&bar[1], 1u, __ATOMIC_RELEASE, __HIP_MEMORY_SCOPE_AGENT);
    } else {
      while (__hip_atomic_load(&bar[1], __ATOMIC_ACQUIRE, __HIP_MEMORY_SCOPE_AGENT) == gen) {
        __builtin_amdgcn_s_sleep(2);
      }
    }
  }
  __syncthreads();
}

// softmax over 128 cols per batch row; 8 threads/row x 16 cols each.
// Writes feedback sm (bf16) into smb; optionally fp32 probs to out row.
__device__ __forceinline__ void softmax128(float* gates, float* red, __bf16* smb,
                                           float* outrow, int tid)
{
  int m = tid >> 3, s = tid & 7;
  int cb = s * 16;
  float mx = -3.0e38f;
  #pragma unroll
  for (int c = 0; c < 16; ++c) mx = fmaxf(mx, gates[m * GST + cb + c]);
  red[m * 8 + s] = mx;
  __syncthreads();
  float rowmax = red[m * 8 + 0];
  #pragma unroll
  for (int k = 1; k < 8; ++k) rowmax = fmaxf(rowmax, red[m * 8 + k]);
  __syncthreads();
  float sum = 0.f;
  #pragma unroll
  for (int c = 0; c < 16; ++c) {
    float e = __expf(gates[m * GST + cb + c] - rowmax);
    gates[m * GST + cb + c] = e;
    sum += e;
  }
  red[m * 8 + s] = sum;
  __syncthreads();
  float tot = 0.f;
  #pragma unroll
  for (int k = 0; k < 8; ++k) tot += red[m * 8 + k];
  float inv = 1.f / tot;
  #pragma unroll
  for (int c = 0; c < 16; ++c) {
    float p = gates[m * GST + cb + c] * inv;
    smb[(size_t)m * SST + cb + c] = (__bf16)p;
    if (outrow) outrow[(size_t)m * Tz * Kz + cb + c] = p;
  }
  __syncthreads();
}

// Issue per-thread async global->LDS copies of one timestep's x into an LDS
// x-buffer (ASYNCcnt-tracked; consumer does s_wait_asynccnt + barrier).
__device__ __forceinline__ void async_stage_x(const __bf16* __restrict__ hbf, int t,
                                              __bf16* xdst, const char* smem, int tid)
{
  for (int idx = tid; idx < Bz * (Dz / 8); idx += TPB) {
    int r = idx >> 7, c = idx & 127;
    unsigned lo = (unsigned)((const char*)(xdst + (size_t)r * XST + c * 8) - smem);
    unsigned long long ga =
        (unsigned long long)(unsigned long long)(uintptr_t)(hbf + ((size_t)r * Tz + t) * Dz + c * 8);
    asm volatile("global_load_async_to_lds_b128 %0, %1, off"
                 :: "v"(lo), "v"(ga) : "memory");
  }
}

// ---------------------------------------------------------------------------
// Persistent LSTM kernel: 16 WGs x 256 threads (8 wave32 each).
// WG w owns hidden units [w*32, w*32+32) of both LSTM layers (all 4 gates).
// ---------------------------------------------------------------------------
__global__ void __launch_bounds__(TPB, 1) lstm_tag_kernel(
    const __bf16* __restrict__ hbf,
    const __bf16* __restrict__ Wih0p, const __bf16* __restrict__ Whh0p,
    const __bf16* __restrict__ Wih1p, const __bf16* __restrict__ Whh1p,
    const __bf16* __restrict__ Woutp,
    const float* __restrict__ b_ih0, const float* __restrict__ b_hh0,
    const float* __restrict__ b_ih1, const float* __restrict__ b_hh1,
    const float* __restrict__ b_out,
    const float* __restrict__ h_t, const float* __restrict__ c_t,
    __bf16* __restrict__ gH1, __bf16* __restrict__ gH2,
    unsigned* __restrict__ bar, float* __restrict__ out)
{
  extern __shared__ char smem[];
  __bf16* xb0   = (__bf16*)smem;                 // [32][XST] x double buffer 0
  __bf16* xb1   = xb0 + 32 * XST;                // [32][XST] x double buffer 1
  __bf16* smb   = xb1 + 32 * XST;                // [32][SST] softmax feedback
  __bf16* h1b   = smb + 32 * SST;                // [32][HST]
  __bf16* h2b   = h1b + 32 * HST;                // [32][HST]
  float*  gates = (float*)(h2b + 32 * HST);      // [32][GST]
  float*  c1b   = gates + 32 * GST;              // [32][32]
  float*  c2b   = c1b + 1024;                    // [32][32]
  float*  red   = c2b + 1024;                    // [32][8]

  const int tid  = threadIdx.x;
  const int lane = tid & 31;
  const int wv   = tid >> 5;        // 0..7
  const int wg   = blockIdx.x;      // 0..15
  const int l15  = lane & 15;

  // Gate column this wave owns (same indexing for both LSTM layers):
  // gate gi = wv>>1 (i,f,g,o blocks of 512), sub-tile = wv&1.
  const int n01 = (wv >> 1) * 512 + wg * 32 + (wv & 1) * 16 + l15;
  const int ntile01 = n01 >> 4;
  const float bias0 = b_ih0[n01] + b_hh0[n01];
  const float bias1 = b_ih1[n01] + b_hh1[n01];
  const float biasL = b_out[wv * 16 + l15];

  // ---- Prologue: init h/c state, initial sm from hiddens[:,0,:H] ----
  for (int idx = tid; idx < Bz * Hz; idx += TPB) {
    int r = idx >> 9, c = idx & 511;
    h1b[r * HST + c] = (__bf16)h_t[idx];
    h2b[r * HST + c] = (__bf16)h_t[Bz * Hz + idx];
  }
  for (int idx = tid; idx < 1024; idx += TPB) {
    int m = idx >> 5, j = idx & 31;
    c1b[idx] = c_t[m * Hz + wg * 32 + j];
    c2b[idx] = c_t[Bz * Hz + m * Hz + wg * 32 + j];
  }
  // hiddens[:,0,:512] (bf16) -> xb0[:,0:512] as A matrix for initial logits
  for (int idx = tid; idx < Bz * (Hz / 8); idx += TPB) {
    int r = idx >> 6, c = idx & 63;
    *((int4*)(xb0 + (size_t)r * XST) + c) =
        *((const int4*)(hbf + (size_t)r * Tz * Dz) + c);
  }
  __syncthreads();
  {
    v8f a0, a1;
    #pragma unroll
    for (int v = 0; v < 8; ++v) { a0[v] = biasL; a1[v] = biasL; }
    wave_gemm(Woutp, wv, Hz / 32, 0, Hz / 32, xb0, XST, lane, a0, a1);
    store_tile(gates, wv * 16, a0, 0, lane);
    store_tile(gates, wv * 16, a1, 1, lane);
  }
  __syncthreads();
  softmax128(gates, red, smb, nullptr, tid);   // sm_0 -> smb

  // Prefetch x_0 into buffer 0 asynchronously (waited at loop top).
  async_stage_x(hbf, 0, xb0, smem, tid);

  // ---- Time loop ----
  for (int t = 0; t < Tz; ++t) {
    __bf16* xcur = (t & 1) ? xb1 : xb0;
    __bf16* xnxt = (t & 1) ? xb0 : xb1;

    // x_t async copy must have landed in LDS before anyone reads it.
    asm volatile("s_wait_asynccnt 0x0" ::: "memory");
    __syncthreads();

    // Layer 0: gates = [x_t, sm] @ W_ih0^T + h1 @ W_hh0^T + bias
    {
      v8f a0, a1;
      #pragma unroll
      for (int v = 0; v < 8; ++v) { a0[v] = bias0; a1[v] = bias0; }
      wave_gemm(Wih0p, ntile01, DKz / 32, 0,       Dz / 32, xcur, XST, lane, a0, a1);
      wave_gemm(Wih0p, ntile01, DKz / 32, Dz / 32, Kz / 32, smb,  SST, lane, a0, a1);
      wave_gemm(Whh0p, ntile01, Hz / 32,  0,       Hz / 32, h1b,  HST, lane, a0, a1);
      store_tile(gates, wv * 16, a0, 0, lane);
      store_tile(gates, wv * 16, a1, 1, lane);
    }
    __syncthreads();

    // x_t fully consumed: start async staging of x_{t+1} into the other buffer;
    // it overlaps pointwise + grid barrier + layer 1 + logits + softmax.
    if (t + 1 < Tz) async_stage_x(hbf, t + 1, xnxt, smem, tid);

    // pointwise LSTM cell 0: update c1 slice, publish h1 slice to L2 scratch
    for (int e = 0; e < 4; ++e) {
      int idx = tid + e * TPB; int m = idx >> 5, j = idx & 31;
      float gi = sigf(gates[m * GST + j]);
      float gf = sigf(gates[m * GST + 32 + j]);
      float gg = tanhf(gates[m * GST + 64 + j]);
      float go = sigf(gates[m * GST + 96 + j]);
      float c = gf * c1b[idx] + gi * gg;
      c1b[idx] = c;
      gH1[(size_t)m * Hz + wg * 32 + j] = (__bf16)(go * tanhf(c));
    }
    grid_barrier(bar);

    // gather full h1 into LDS
    for (int idx = tid; idx < Bz * (Hz / 8); idx += TPB) {
      int r = idx >> 6, c = idx & 63;
      *((int4*)(h1b + (size_t)r * HST) + c) =
          *((const int4*)(gH1 + (size_t)r * Hz) + c);
    }
    __syncthreads();

    // Layer 1: gates = h1 @ W_ih1^T + h2 @ W_hh1^T + bias
    {
      v8f a0, a1;
      #pragma unroll
      for (int v = 0; v < 8; ++v) { a0[v] = bias1; a1[v] = bias1; }
      wave_gemm(Wih1p, ntile01, Hz / 32, 0, Hz / 32, h1b, HST, lane, a0, a1);
      wave_gemm(Whh1p, ntile01, Hz / 32, 0, Hz / 32, h2b, HST, lane, a0, a1);
      store_tile(gates, wv * 16, a0, 0, lane);
      store_tile(gates, wv * 16, a1, 1, lane);
    }
    __syncthreads();

    for (int e = 0; e < 4; ++e) {
      int idx = tid + e * TPB; int m = idx >> 5, j = idx & 31;
      float gi = sigf(gates[m * GST + j]);
      float gf = sigf(gates[m * GST + 32 + j]);
      float gg = tanhf(gates[m * GST + 64 + j]);
      float go = sigf(gates[m * GST + 96 + j]);
      float c = gf * c2b[idx] + gi * gg;
      c2b[idx] = c;
      gH2[(size_t)m * Hz + wg * 32 + j] = (__bf16)(go * tanhf(c));
    }
    grid_barrier(bar);

    for (int idx = tid; idx < Bz * (Hz / 8); idx += TPB) {
      int r = idx >> 6, c = idx & 63;
      *((int4*)(h2b + (size_t)r * HST) + c) =
          *((const int4*)(gH2 + (size_t)r * Hz) + c);
    }
    __syncthreads();

    // logits = h2 @ W_out^T + b_out (redundant per WG -> no 3rd barrier)
    {
      v8f a0, a1;
      #pragma unroll
      for (int v = 0; v < 8; ++v) { a0[v] = biasL; a1[v] = biasL; }
      wave_gemm(Woutp, wv, Hz / 32, 0, Hz / 32, h2b, HST, lane, a0, a1);
      store_tile(gates, wv * 16, a0, 0, lane);
      store_tile(gates, wv * 16, a1, 1, lane);
    }
    __syncthreads();
    // mask shift is per-row constant => softmax(logits+shift)==softmax(logits)
    softmax128(gates, red, smb, (wg == 0) ? (out + (size_t)t * Kz) : nullptr, tid);
  }
}

// ---------------------------------------------------------------------------
// Prologue kernels
// ---------------------------------------------------------------------------
__global__ void cvt_bf16_kernel(const float* __restrict__ src,
                                __bf16* __restrict__ dst, long long n)
{
  long long i = (long long)blockIdx.x * blockDim.x + threadIdx.x;
  long long stride = (long long)gridDim.x * blockDim.x;
  for (; i < n; i += stride) dst[i] = (__bf16)src[i];
}

// Repack W [N][Kdim] f32 row-major -> bf16 WMMA B fragments:
// dst[((nt*KT + kt)*32 + l)*16 + e] = W[nt*16 + (l&15)][kt*32 + ((l>>4)&1)*16 + e]
__global__ void pack_w_kernel(const float* __restrict__ W,
                              __bf16* __restrict__ dst, int N, int Kdim)
{
  long long total = (long long)N * Kdim;
  int KT = Kdim >> 5;
  long long i = (long long)blockIdx.x * blockDim.x + threadIdx.x;
  long long stride = (long long)gridDim.x * blockDim.x;
  for (; i < total; i += stride) {
    int e = (int)(i & 15);
    int l = (int)((i >> 4) & 31);
    long long tile = i >> 9;
    int kt = (int)(tile % KT);
    int nt = (int)(tile / KT);
    int n = nt * 16 + (l & 15);
    int k = kt * 32 + ((l >> 4) & 1) * 16 + e;
    dst[i] = (__bf16)W[(long long)n * Kdim + k];
  }
}

__global__ void init_bar_kernel(unsigned* bar)
{
  if (threadIdx.x < 8) bar[threadIdx.x] = 0u;
}

// ---------------------------------------------------------------------------
extern "C" void kernel_launch(void* const* d_in, const int* in_sizes, int n_in,
                              void* d_out, int out_size, void* d_ws, size_t ws_size,
                              hipStream_t stream)
{
  (void)in_sizes; (void)n_in; (void)out_size; (void)ws_size;
  const float* hiddens = (const float*)d_in[0];
  const float* h_t     = (const float*)d_in[1];
  const float* c_t     = (const float*)d_in[2];
  const float* mask    = (const float*)d_in[3]; (void)mask;  // softmax-invariant shift
  const float* W_out   = (const float*)d_in[4];
  const float* b_out   = (const float*)d_in[5];
  const float* W_ih0   = (const float*)d_in[6];
  const float* W_hh0   = (const float*)d_in[7];
  const float* b_ih0   = (const float*)d_in[8];
  const float* b_hh0   = (const float*)d_in[9];
  const float* W_ih1   = (const float*)d_in[10];
  const float* W_hh1   = (const float*)d_in[11];
  const float* b_ih1   = (const float*)d_in[12];
  const float* b_hh1   = (const float*)d_in[13];

  char* ws = (char*)d_ws;
  size_t off = 0;
  auto alloc = [&](size_t bytes) -> char* {
    char* p = ws + off;
    off += (bytes + 255) & ~(size_t)255;
    return p;
  };
  unsigned* bar  = (unsigned*)alloc(256);
  __bf16* gH1    = (__bf16*)alloc((size_t)Bz * Hz * 2);
  __bf16* gH2    = (__bf16*)alloc((size_t)Bz * Hz * 2);
  __bf16* hbf    = (__bf16*)alloc((size_t)Bz * Tz * Dz * 2);
  __bf16* Wih0p  = (__bf16*)alloc((size_t)G4z * DKz * 2);
  __bf16* Whh0p  = (__bf16*)alloc((size_t)G4z * Hz * 2);
  __bf16* Wih1p  = (__bf16*)alloc((size_t)G4z * Hz * 2);
  __bf16* Whh1p  = (__bf16*)alloc((size_t)G4z * Hz * 2);
  __bf16* Woutp  = (__bf16*)alloc((size_t)Kz * Hz * 2);

  cvt_bf16_kernel<<<2048, 256, 0, stream>>>(hiddens, hbf, (long long)Bz * Tz * Dz);
  pack_w_kernel<<<1024, 256, 0, stream>>>(W_ih0, Wih0p, G4z, DKz);
  pack_w_kernel<<<1024, 256, 0, stream>>>(W_hh0, Whh0p, G4z, Hz);
  pack_w_kernel<<<1024, 256, 0, stream>>>(W_ih1, Wih1p, G4z, Hz);
  pack_w_kernel<<<1024, 256, 0, stream>>>(W_hh1, Whh1p, G4z, Hz);
  pack_w_kernel<<<256, 256, 0, stream>>>(W_out, Woutp, Kz, Hz);
  init_bar_kernel<<<1, 32, 0, stream>>>(bar);

  lstm_tag_kernel<<<NWG, TPB, LDS_BYTES, stream>>>(
      hbf, Wih0p, Whh0p, Wih1p, Whh1p, Woutp,
      b_ih0, b_hh0, b_ih1, b_hh1, b_out,
      h_t, c_t, gH1, gH2, bar, (float*)d_out);
}
